// SelfAttention_16853451669845
// MI455X (gfx1250) — matrix operand order repaired and checked by reading, not compiled, not run on hardware
//
#include <hip/hip_runtime.h>

// ---------------------------------------------------------------------------
// Causal self-attention for MI455X (gfx1250, wave32, WMMA bf16 16x16x32).
//   0) X -> bf16; W{q,k,v} -> bf16 transposed   (one-time, bandwidth-trivial)
//   1) Q,K = Xb @ Wt (bf16), V stored TRANSPOSED; 1/sqrt(d) folded into Q
//   2) S = Qs @ K^T   (lower-triangle tiles only, -inf above diagonal)
//   3) row softmax    (f32 math, in-place bf16 P)
//   4) O = P @ V      (causal k-range, f32 out)
// All GEMM staging is row-major 16B copies -> GLOBAL_LOAD_ASYNC_TO_LDS_B128.
// ---------------------------------------------------------------------------

typedef __bf16 bf16;
typedef __attribute__((ext_vector_type(16))) __bf16 v16bf;
typedef __attribute__((ext_vector_type(8)))  __bf16 v8bf;
typedef __attribute__((ext_vector_type(4)))  __bf16 v4bf;
typedef __attribute__((ext_vector_type(8)))  float  v8f;
typedef __attribute__((ext_vector_type(4)))  int    v4i;

#define N_TOK   4096
#define EMBED   2048
#define D_ATTN  2048
#define LSTR    72          // LDS row stride in bf16 (144B: 16B aligned, banks spread)
#define KSTEP   64          // k-elements staged per LDS tile (2 WMMA k-steps)

// ---- async memory->LDS copy (CDNA5), with portable fallback ---------------
#if defined(__has_builtin)
#  if __has_builtin(__builtin_amdgcn_global_load_async_to_lds_b128)
#    define HAVE_ASYNC_LDS 1
#  endif
#endif
#ifndef HAVE_ASYNC_LDS
#  define HAVE_ASYNC_LDS 0
#endif

__device__ __forceinline__ void cp16(bf16* dst_lds, const bf16* src_glb) {
#if HAVE_ASYNC_LDS
    __builtin_amdgcn_global_load_async_to_lds_b128(
        (__attribute__((address_space(1))) v4i*)(src_glb),
        (__attribute__((address_space(3))) v4i*)(dst_lds), 0, 0);
#else
    *(v8bf*)dst_lds = *(const v8bf*)src_glb;
#endif
}

#if HAVE_ASYNC_LDS
#  if __has_builtin(__builtin_amdgcn_s_wait_asynccnt)
#    define WAIT_ASYNC() __builtin_amdgcn_s_wait_asynccnt(0)
#  else
#    define WAIT_ASYNC() asm volatile("s_wait_asynccnt 0x0" ::: "memory")
#  endif
#else
#  define WAIT_ASYNC()
#endif

// ---- WMMA fragment gathers (ISA 7.12.2 layouts) ---------------------------
// A (16x32 bf16): lane L -> row L%16, k = {h*8..h*8+7} u {16+h*8..}, h=L/16
__device__ __forceinline__ v16bf frag_a(const bf16* lds, int row, int koff) {
    union { v16bf v; v8bf h[2]; } u;
    const bf16* p = lds + row * LSTR + koff;
    u.h[0] = *(const v8bf*)(p);
    u.h[1] = *(const v8bf*)(p + 16);
    return u.v;
}
// B (32x16 bf16): lane L -> col L%16, k = {h*8..h*8+7} u {16+h*8..}, h=L/16,
// read from an N-major (pre-transposed) LDS tile.
__device__ __forceinline__ v16bf frag_b(const bf16* lds, int col, int kbase) {
    union { v16bf v; v8bf h[2]; } u;
    const bf16* p = lds + col * LSTR + kbase;
    u.h[0] = *(const v8bf*)(p);
    u.h[1] = *(const v8bf*)(p + 16);
    return u.v;
}

// Each wave computes a 64x32 patch as 4x2 grid of 16x16 WMMA accumulators.
__device__ __forceinline__ void mma_tile(const bf16* As, const bf16* Bs,
                                         int wm, int wc, int half, int l16,
                                         v8f acc[4][2]) {
#pragma unroll
    for (int kk = 0; kk < KSTEP; kk += 32) {
        v16bf a[4], b[2];
#pragma unroll
        for (int mi = 0; mi < 4; ++mi)
            a[mi] = frag_a(As, wm * 64 + mi * 16 + l16, kk + half * 8);
#pragma unroll
        for (int ni = 0; ni < 2; ++ni)
            b[ni] = frag_b(Bs, wc * 32 + ni * 16 + l16, kk + half * 8);
#pragma unroll
        for (int mi = 0; mi < 4; ++mi)
#pragma unroll
            for (int ni = 0; ni < 2; ++ni)
                acc[mi][ni] = __builtin_amdgcn_wmma_f32_16x16x32_bf16(
                    false, a[mi], false, b[ni], (short)0, acc[mi][ni], false, false);
    }
}

// ---------------- Kernel 0a: X f32 -> bf16 ---------------------------------
__global__ __launch_bounds__(256) void convert_x(const float* __restrict__ X,
                                                 bf16* __restrict__ Xb) {
    const size_t base = ((size_t)blockIdx.x * 256 + threadIdx.x) * 8;
    float4 f0 = *(const float4*)(X + base);
    float4 f1 = *(const float4*)(X + base + 4);
    v8bf o;
    o[0] = (bf16)f0.x; o[1] = (bf16)f0.y; o[2] = (bf16)f0.z; o[3] = (bf16)f0.w;
    o[4] = (bf16)f1.x; o[5] = (bf16)f1.y; o[6] = (bf16)f1.z; o[7] = (bf16)f1.w;
    *(v8bf*)(Xb + base) = o;
}

// ---------------- Kernel 0b: W f32 -> bf16, transposed ---------------------
__global__ __launch_bounds__(256) void transpose_w(
    const float* __restrict__ Wq, const float* __restrict__ Wk,
    const float* __restrict__ Wv,
    bf16* __restrict__ Wtq, bf16* __restrict__ Wtk, bf16* __restrict__ Wtv) {
    __shared__ __align__(16) bf16 T[64 * LSTR];
    const int z = blockIdx.z;
    const float* W  = (z == 0) ? Wq  : (z == 1) ? Wk  : Wv;
    bf16*        Wt = (z == 0) ? Wtq : (z == 1) ? Wtk : Wtv;
    const int k0 = blockIdx.y * 64, n0 = blockIdx.x * 64;
    const int tid = threadIdx.x;
#pragma unroll
    for (int i = 0; i < 4; ++i) {              // load 64x64 tile, convert
        int v = tid + i * 256;
        int r = v >> 4, c4 = v & 15;
        float4 f = *(const float4*)(W + (size_t)(k0 + r) * D_ATTN + n0 + c4 * 4);
        v4bf o; o[0] = (bf16)f.x; o[1] = (bf16)f.y; o[2] = (bf16)f.z; o[3] = (bf16)f.w;
        *(v4bf*)&T[r * LSTR + c4 * 4] = o;
    }
    __syncthreads();
#pragma unroll
    for (int i = 0; i < 2; ++i) {              // write transposed, coalesced 16B
        int c = tid + i * 256;
        int n = c >> 3, kc = c & 7;
        v8bf o;
#pragma unroll
        for (int j = 0; j < 8; ++j) o[j] = T[(kc * 8 + j) * LSTR + n];
        *(v8bf*)(Wt + (size_t)(n0 + n) * EMBED + k0 + kc * 8) = o;
    }
}

// ---------------- Kernel 1: Q/K normal, V transposed -----------------------
__global__ __launch_bounds__(256) void qkv_gemm(
    const bf16* __restrict__ Xb,
    const bf16* __restrict__ Wtq, const bf16* __restrict__ Wtk,
    const bf16* __restrict__ Wtv,
    bf16* __restrict__ Qb, bf16* __restrict__ Kb, bf16* __restrict__ Vt) {
    __shared__ __align__(16) bf16 As[128 * LSTR];
    __shared__ __align__(16) bf16 Bs[128 * LSTR];
    const int tid  = threadIdx.x;
    const int wave = tid >> 5, lane = tid & 31;
    const int half = lane >> 4, l16 = lane & 15;
    const int wm = wave >> 2, wc = wave & 3;
    const int m0 = blockIdx.y * 128, n0 = blockIdx.x * 128;
    const int z  = blockIdx.z;
    const bf16* Wt = (z == 0) ? Wtq : (z == 1) ? Wtk : Wtv;
    const float scale = (z == 0) ? 0.02209708691207961f : 1.0f; // 1/sqrt(2048)

    v8f acc[4][2];
#pragma unroll
    for (int mi = 0; mi < 4; ++mi)
#pragma unroll
        for (int ni = 0; ni < 2; ++ni) acc[mi][ni] = (v8f){};

    for (int kt = 0; kt < EMBED / KSTEP; ++kt) {
        __syncthreads();
#pragma unroll
        for (int i = 0; i < 4; ++i) {          // A: Xb rows (async 16B)
            int v = tid + i * 256;
            int r = v >> 3, c8 = v & 7;
            cp16(&As[r * LSTR + c8 * 8],
                 Xb + (size_t)(m0 + r) * EMBED + kt * KSTEP + c8 * 8);
        }
#pragma unroll
        for (int i = 0; i < 4; ++i) {          // B: Wt rows are B^T columns
            int v = tid + i * 256;
            int r = v >> 3, c8 = v & 7;
            cp16(&Bs[r * LSTR + c8 * 8],
                 Wt + (size_t)(n0 + r) * EMBED + kt * KSTEP + c8 * 8);
        }
        WAIT_ASYNC();
        __syncthreads();
        mma_tile(As, Bs, wm, wc, half, l16, acc);
    }

    if (z == 2) {                              // V: store transposed (16B/lane)
#pragma unroll
        for (int mi = 0; mi < 4; ++mi)
#pragma unroll
            for (int ni = 0; ni < 2; ++ni) {
                int gc  = n0 + wc * 32 + ni * 16 + l16;
                int gr0 = m0 + wm * 64 + mi * 16 + half * 8;
                v8bf o;
#pragma unroll
                for (int e = 0; e < 8; ++e) o[e] = (bf16)acc[mi][ni][e];
                *(v8bf*)(Vt + (size_t)gc * N_TOK + gr0) = o;
            }
    } else {
        bf16* Out = (z == 0) ? Qb : Kb;
#pragma unroll
        for (int mi = 0; mi < 4; ++mi)
#pragma unroll
            for (int ni = 0; ni < 2; ++ni)
#pragma unroll
                for (int e = 0; e < 8; ++e) {
                    int gr = m0 + wm * 64 + mi * 16 + half * 8 + e;
                    int gc = n0 + wc * 32 + ni * 16 + l16;
                    Out[(size_t)gr * D_ATTN + gc] = (bf16)(acc[mi][ni][e] * scale);
                }
    }
}

// ---------------- Kernel 2: S = Qs @ K^T (causal tiles only) ---------------
__global__ __launch_bounds__(256) void scores_gemm(
    const bf16* __restrict__ Qb, const bf16* __restrict__ Kb,
    bf16* __restrict__ S) {
    const int tm = blockIdx.y, tn = blockIdx.x;
    if (tn > tm) return;
    __shared__ __align__(16) bf16 As[128 * LSTR];
    __shared__ __align__(16) bf16 Bs[128 * LSTR];
    const int tid  = threadIdx.x;
    const int wave = tid >> 5, lane = tid & 31;
    const int half = lane >> 4, l16 = lane & 15;
    const int wm = wave >> 2, wc = wave & 3;
    const int m0 = tm * 128, n0 = tn * 128;

    v8f acc[4][2];
#pragma unroll
    for (int mi = 0; mi < 4; ++mi)
#pragma unroll
        for (int ni = 0; ni < 2; ++ni) acc[mi][ni] = (v8f){};

    for (int kt = 0; kt < D_ATTN / KSTEP; ++kt) {
        __syncthreads();
#pragma unroll
        for (int i = 0; i < 4; ++i) {          // A: Q rows
            int v = tid + i * 256;
            int r = v >> 3, c8 = v & 7;
            cp16(&As[r * LSTR + c8 * 8],
                 Qb + (size_t)(m0 + r) * D_ATTN + kt * KSTEP + c8 * 8);
        }
#pragma unroll
        for (int i = 0; i < 4; ++i) {          // B: K rows ARE K^T columns
            int v = tid + i * 256;
            int r = v >> 3, c8 = v & 7;
            cp16(&Bs[r * LSTR + c8 * 8],
                 Kb + (size_t)(n0 + r) * D_ATTN + kt * KSTEP + c8 * 8);
        }
        WAIT_ASYNC();
        __syncthreads();
        mma_tile(As, Bs, wm, wc, half, l16, acc);
    }
    const float ninf = -__builtin_inff();
#pragma unroll
    for (int mi = 0; mi < 4; ++mi)
#pragma unroll
        for (int ni = 0; ni < 2; ++ni)
#pragma unroll
            for (int e = 0; e < 8; ++e) {
                int gr = m0 + wm * 64 + mi * 16 + half * 8 + e;
                int gc = n0 + wc * 32 + ni * 16 + l16;
                float val = (gc > gr) ? ninf : acc[mi][ni][e];  // causal mask
                S[(size_t)gr * N_TOK + gc] = (bf16)val;
            }
}

// ---------------- Kernel 3: per-row softmax, in place ----------------------
__global__ __launch_bounds__(256) void softmax_rows(bf16* __restrict__ S) {
    const int row = blockIdx.x;
    const int L   = ((row >> 7) + 1) << 7;     // written (causal) extent
    bf16* p = S + (size_t)row * N_TOK;
    const int tid = threadIdx.x;
    __shared__ float red[256];

    float m = -__builtin_inff();
    for (int j = tid; j < L; j += 256) m = fmaxf(m, (float)p[j]);
    red[tid] = m; __syncthreads();
    for (int s = 128; s > 0; s >>= 1) {
        if (tid < s) red[tid] = fmaxf(red[tid], red[tid + s]);
        __syncthreads();
    }
    m = red[0]; __syncthreads();

    float sum = 0.f;
    for (int j = tid; j < L; j += 256) sum += __expf((float)p[j] - m);
    red[tid] = sum; __syncthreads();
    for (int s = 128; s > 0; s >>= 1) {
        if (tid < s) red[tid] += red[tid + s];
        __syncthreads();
    }
    const float inv = 1.0f / red[0];
    for (int j = tid; j < L; j += 256)
        p[j] = (bf16)(__expf((float)p[j] - m) * inv);   // exp(-inf - m) = 0
}

// ---------------- Kernel 4: O = P @ V (causal k-range, f32 out) ------------
__global__ __launch_bounds__(256) void out_gemm(
    const bf16* __restrict__ P, const bf16* __restrict__ Vt,
    float* __restrict__ O) {
    __shared__ __align__(16) bf16 As[128 * LSTR];
    __shared__ __align__(16) bf16 Bs[128 * LSTR];
    const int tid  = threadIdx.x;
    const int wave = tid >> 5, lane = tid & 31;
    const int half = lane >> 4, l16 = lane & 15;
    const int wm = wave >> 2, wc = wave & 3;
    const int m0 = blockIdx.y * 128, n0 = blockIdx.x * 128;
    const int nkt = (blockIdx.y + 1) * (128 / KSTEP);   // causal: keys <= queries

    v8f acc[4][2];
#pragma unroll
    for (int mi = 0; mi < 4; ++mi)
#pragma unroll
        for (int ni = 0; ni < 2; ++ni) acc[mi][ni] = (v8f){};

    for (int kt = 0; kt < nkt; ++kt) {
        __syncthreads();
#pragma unroll
        for (int i = 0; i < 4; ++i) {          // A: P rows (stride N_TOK)
            int v = tid + i * 256;
            int r = v >> 3, c8 = v & 7;
            cp16(&As[r * LSTR + c8 * 8],
                 P + (size_t)(m0 + r) * N_TOK + kt * KSTEP + c8 * 8);
        }
#pragma unroll
        for (int i = 0; i < 4; ++i) {          // B: Vt rows are V columns
            int v = tid + i * 256;
            int r = v >> 3, c8 = v & 7;
            cp16(&Bs[r * LSTR + c8 * 8],
                 Vt + (size_t)(n0 + r) * N_TOK + kt * KSTEP + c8 * 8);
        }
        WAIT_ASYNC();
        __syncthreads();
        mma_tile(As, Bs, wm, wc, half, l16, acc);
    }
#pragma unroll
    for (int mi = 0; mi < 4; ++mi)
#pragma unroll
        for (int ni = 0; ni < 2; ++ni)
#pragma unroll
            for (int e = 0; e < 8; ++e) {
                int gr = m0 + wm * 64 + mi * 16 + half * 8 + e;
                int gc = n0 + wc * 32 + ni * 16 + l16;
                O[(size_t)gr * D_ATTN + gc] = acc[mi][ni][e];
            }
}

// ---------------------------------------------------------------------------
extern "C" void kernel_launch(void* const* d_in, const int* in_sizes, int n_in,
                              void* d_out, int out_size, void* d_ws, size_t ws_size,
                              hipStream_t stream) {
    const float* X  = (const float*)d_in[0];
    const float* Wq = (const float*)d_in[1];
    const float* Wk = (const float*)d_in[2];
    const float* Wv = (const float*)d_in[3];
    float* O = (float*)d_out;

    const size_t MiB = 1048576;
    char* ws = (char*)d_ws;
    bf16* Qb  = (bf16*)(ws);                 // 16 MiB  4096x2048 (pre-scaled by 1/sqrt(d))
    bf16* Kb  = (bf16*)(ws + 16 * MiB);      // 16 MiB  4096x2048
    bf16* Vt  = (bf16*)(ws + 32 * MiB);      // 16 MiB  2048x4096 (V transposed)
    bf16* S   = (bf16*)(ws + 48 * MiB);      // 32 MiB  4096x4096 scores -> P
    // Aliased into the S region (dead before S is written; stream-ordered):
    bf16* Xb  = (bf16*)(ws + 48 * MiB);      // 16 MiB  4096x2048 bf16 X
    bf16* Wtq = (bf16*)(ws + 64 * MiB);      //  8 MiB  2048x2048 bf16 Wq^T
    bf16* Wtk = (bf16*)(ws + 72 * MiB);      //  8 MiB  (inside S tail)
    bf16* Wtv = (bf16*)(ws + 80 * MiB);      //  8 MiB  (beyond S) -> 88 MiB total

    convert_x  <<<dim3(N_TOK * EMBED / (256 * 8)), 256, 0, stream>>>(X, Xb);
    transpose_w<<<dim3(D_ATTN / 64, EMBED / 64, 3), 256, 0, stream>>>(Wq, Wk, Wv, Wtq, Wtk, Wtv);
    qkv_gemm   <<<dim3(D_ATTN / 128, N_TOK / 128, 3), 256, 0, stream>>>(Xb, Wtq, Wtk, Wtv, Qb, Kb, Vt);
    scores_gemm<<<dim3(N_TOK / 128, N_TOK / 128),     256, 0, stream>>>(Qb, Kb, S);
    softmax_rows<<<N_TOK,                             256, 0, stream>>>(S);
    out_gemm   <<<dim3(D_ATTN / 128, N_TOK / 128),    256, 0, stream>>>(S, Vt, O);
}
